// MultiheadIsotropicRFA_77103252898105
// MI455X (gfx1250) — matrix operand order repaired and checked by reading, not compiled, not run on hardware
//
#include <hip/hip_runtime.h>
#include <hip/hip_fp16.h>

typedef __attribute__((ext_vector_type(16))) _Float16 v16h;
typedef __attribute__((ext_vector_type(8)))  _Float16 v8h;
typedef __attribute__((ext_vector_type(8)))  float    v8f;
typedef __attribute__((ext_vector_type(4)))  unsigned int u32x4;
typedef __attribute__((ext_vector_type(4)))  int      i32x4;
typedef __attribute__((ext_vector_type(8)))  int      i32x8;

#define B_     2
#define N_     1024
#define DE_    1024
#define H_     16
#define DHALF_ 32
#define DH_    64        // complex dims per head
#define DK_    1024      // H * DH
#define TWODK_ 2048
#define FD_    128       // 2*DH f16 features per head (re||im)
#define JCH_   32        // attention j-chunk

static __device__ __forceinline__ v16h pack16(v8h lo, v8h hi) {
  v16h r;
#pragma unroll
  for (int i = 0; i < 8; ++i) { r[i] = lo[i]; r[8 + i] = hi[i]; }
  return r;
}

// A fragment (16x32 f16), contiguous k = 16*g + s; tile stored row-major [16][32].
static __device__ __forceinline__ v16h frag_a_lds(const _Float16* t) {
  int lane = threadIdx.x & 31;
  int m = lane & 15, g = lane >> 4;
  v8h lo = *(const v8h*)(t + m * 32 + 16 * g);
  v8h hi = *(const v8h*)(t + m * 32 + 16 * g + 8);
  return pack16(lo, hi);
}

// B fragment (32x16 f16); tile stored transposed [16 cols][32 k].
static __device__ __forceinline__ v16h frag_b_lds(const _Float16* t) {
  int lane = threadIdx.x & 31;
  int n = lane & 15, g = lane >> 4;
  return *(const v16h*)(t + n * 32 + 16 * g);
}

static __device__ __forceinline__ v8f wmma_f16(v16h a, v16h b, v8f c) {
  return __builtin_amdgcn_wmma_f32_16x16x32_f16(false, a, false, b, (short)0, c,
                                                false, false);
}

static __device__ __forceinline__ float softplusf(float x) {
  return log1pf(expf(x));
}

static __device__ __forceinline__ unsigned lds_off(const void* p) {
  return (unsigned)(unsigned long long)p;
}

// ---------------------------------------------------------------------------
// TDM 2D tile load: global (f16, row stride stride0 elements) -> LDS packed.
// D# per CDNA5 ISA. data_size=1 (2 bytes).
// ---------------------------------------------------------------------------
static __device__ __forceinline__ void tdm_load_2d(
    const void* gaddr, unsigned ldsaddr, unsigned tensor_d0, unsigned tensor_d1,
    unsigned tile_d0, unsigned tile_d1, unsigned long long stride0) {
  unsigned long long ga = (unsigned long long)gaddr;
  u32x4 g0;
  g0[0] = 1u;                                        // count=1, user-mode D#
  g0[1] = ldsaddr;                                   // lds byte address
  g0[2] = (unsigned)(ga & 0xffffffffu);              // global_addr[31:0]
  g0[3] = (unsigned)((ga >> 32) & 0x1ffffffu) | (2u << 30);  // addr[56:32]|type=2
  i32x8 g1;
  g1[0] = (int)(1u << 16);                           // wg_mask=0, data_size=2B
  g1[1] = (int)((tensor_d0 & 0xffffu) << 16);        // dim0[15:0]
  g1[2] = (int)((tensor_d0 >> 16) | ((tensor_d1 & 0xffffu) << 16));
  g1[3] = (int)((tensor_d1 >> 16) | ((tile_d0 & 0xffffu) << 16));
  g1[4] = (int)(tile_d1 & 0xffffu);                  // tile_dim1, tile_dim2=0
  g1[5] = (int)(stride0 & 0xffffffffu);              // dim0_stride[31:0]
  g1[6] = (int)((unsigned)(stride0 >> 32) & 0xffffu);
  g1[7] = 0;
  i32x4 gz = {0, 0, 0, 0};
#if defined(__clang_major__) && __clang_major__ >= 23
  i32x8 gz8 = {0, 0, 0, 0, 0, 0, 0, 0};
  __builtin_amdgcn_tensor_load_to_lds(g0, g1, gz, gz, gz8, 0);
#else
  __builtin_amdgcn_tensor_load_to_lds(g0, g1, gz, gz, 0);
#endif
}

// ---------------------------------------------------------------------------
// Fused projection + complex rotation (q/k rotate, v transposed store).
// ---------------------------------------------------------------------------
__global__ void __launch_bounds__(128) proj_rot_kernel(
    const float* __restrict__ X, const float* __restrict__ W,
    const float* __restrict__ omega, _Float16* __restrict__ out, int mode) {
  __shared__ __align__(64) _Float16 lds[4][1536];
  int wave = threadIdx.x >> 5, lane = threadIdx.x & 31;
  int tile = blockIdx.x * 4 + wave;          // 8192 tiles
  int row0 = (tile >> 6) * 16;
  int c0   = (tile & 63) * 16;
  _Float16* la  = &lds[wave][0];
  _Float16* lbr = &lds[wave][512];
  _Float16* lbi = &lds[wave][1024];
  int n = lane & 15, g = lane >> 4;
  v8f zero = {};
  v8f acc_re = zero, acc_im = zero;

  for (int k0 = 0; k0 < DE_; k0 += 32) {
    {
      int m = lane >> 1, hsel = lane & 1;
      const float4* p =
          (const float4*)(X + (size_t)(row0 + m) * DE_ + k0 + 16 * hsel);
      if (k0 + 32 < DE_)
        __builtin_prefetch(X + (size_t)(row0 + m) * DE_ + k0 + 32, 0, 1);
#pragma unroll
      for (int q = 0; q < 4; ++q) {
        float4 v = p[q];
        _Float16* d = la + m * 32 + 16 * hsel + 4 * q;
        d[0] = (_Float16)v.x; d[1] = (_Float16)v.y;
        d[2] = (_Float16)v.z; d[3] = (_Float16)v.w;
      }
    }
    {
      const float4* pr = (const float4*)(W + (size_t)(k0 + lane) * TWODK_ + c0);
      const float4* pi =
          (const float4*)(W + (size_t)(k0 + lane) * TWODK_ + DK_ + c0);
#pragma unroll
      for (int q = 0; q < 4; ++q) {
        float4 vr = pr[q], vi = pi[q];
        lbr[(4 * q + 0) * 32 + lane] = (_Float16)vr.x;
        lbr[(4 * q + 1) * 32 + lane] = (_Float16)vr.y;
        lbr[(4 * q + 2) * 32 + lane] = (_Float16)vr.z;
        lbr[(4 * q + 3) * 32 + lane] = (_Float16)vr.w;
        lbi[(4 * q + 0) * 32 + lane] = (_Float16)vi.x;
        lbi[(4 * q + 1) * 32 + lane] = (_Float16)vi.y;
        lbi[(4 * q + 2) * 32 + lane] = (_Float16)vi.z;
        lbi[(4 * q + 3) * 32 + lane] = (_Float16)vi.w;
      }
    }
    v16h a  = frag_a_lds(la);
    v16h br = frag_b_lds(lbr);
    v16h bi = frag_b_lds(lbi);
    acc_re = wmma_f16(a, br, acc_re);
    acc_im = wmma_f16(a, bi, acc_im);
  }

  int h  = c0 / DH_;
  int d0 = c0 % DH_;
  int d  = d0 + n;
  float om = (d < DHALF_) ? omega[h * DHALF_ + d] : -omega[h * DHALF_ + d - DHALF_];
#pragma unroll
  for (int r = 0; r < 8; ++r) {
    int grow = row0 + r + 8 * g;
    int b = grow >> 10, t = grow & (N_ - 1);
    float re = acc_re[r], im = acc_im[r];
    float rr, ri;
    if (mode < 2) {
      float ang = om * (float)t;
      float c = cosf(ang), s = sinf(ang);
      rr = re * c - im * s;
      ri = re * s + im * c;
    } else {
      rr = re; ri = im;
    }
    if (mode < 2) {
      size_t base = ((size_t)(b * H_ + h) * N_ + t) * FD_;
      out[base + d]       = (_Float16)rr;
      out[base + DH_ + d] = (_Float16)ri;
    } else {
      size_t base = (size_t)(b * H_ + h) * FD_ * N_;
      out[base + (size_t)d * N_ + t]         = (_Float16)rr;
      out[base + (size_t)(DH_ + d) * N_ + t] = (_Float16)ri;
    }
  }
}

// ---------------------------------------------------------------------------
__global__ void __launch_bounds__(256) norms_kernel(
    const _Float16* __restrict__ Qr, const _Float16* __restrict__ Kr,
    float* __restrict__ qn, float* __restrict__ kn) {
  int idx = blockIdx.x * 256 + threadIdx.x;
  const _Float16* q = Qr + (size_t)idx * FD_;
  const _Float16* k = Kr + (size_t)idx * FD_;
  float sq = 0.f, sk = 0.f;
#pragma unroll
  for (int c = 0; c < 8; ++c) {
    v16h a = *(const v16h*)(q + 16 * c);
    v16h b = *(const v16h*)(k + 16 * c);
#pragma unroll
    for (int i = 0; i < 16; ++i) {
      float fa = (float)a[i]; sq += fa * fa;
      float fb = (float)b[i]; sk += fb * fb;
    }
  }
  qn[idx] = sq; kn[idx] = sk;
}

// ---------------------------------------------------------------------------
// Flash attention: workgroup = 4 waves = 64 i-rows of one (b,h).
// K'/V j-chunks staged in LDS by the TDM, double-buffered, shared by 4 waves.
// ---------------------------------------------------------------------------
__global__ void __launch_bounds__(128) attn_kernel(
    const _Float16* __restrict__ Qr, const _Float16* __restrict__ Kr,
    const _Float16* __restrict__ Vt, const float* __restrict__ qn,
    const float* __restrict__ kn, const float* __restrict__ mu_raw,
    const float* __restrict__ sigma_raw, const float* __restrict__ eta_raw,
    const float* __restrict__ gamma_raw, const float* __restrict__ tau,
    const float* __restrict__ nu_raw, float* __restrict__ attn_out) {
  __shared__ __align__(128) _Float16 lk[2][JCH_ * FD_];  // K chunk [32 tok][128]
  __shared__ __align__(128) _Float16 lv[2][FD_ * JCH_];  // V chunk [128][32 tok]
  __shared__ __align__(64)  _Float16 ldsp[4][512];       // per-wave P tile
  int wave = threadIdx.x >> 5, lane = threadIdx.x & 31;
  int blk = blockIdx.x;           // [0, B*H*(N/64))
  int iblk = blk & 15;            // N/64 = 16
  int bh = blk >> 4;
  int h = bh & (H_ - 1);
  int b = bh >> 4;
  int i0 = iblk * 64 + wave * 16;
  int n = lane & 15, g = lane >> 4;

  float mu = 1.0f / (1.0f + expf(-mu_raw[h])) + 1e-4f;
  float sigma_sq = softplusf(sigma_raw[h]);
  float addv = softplusf(eta_raw[h]) + softplusf(gamma_raw[h]);
  float nu = softplusf(nu_raw[h]);
  float coef = -0.5f * tau[h] * (nu + (float)DH_);
  float sig2mu = sigma_sq / (2.0f * mu);
  float inv_nu = 1.0f / nu;

  const _Float16* qbase = Qr + ((size_t)bh * N_ + i0) * FD_;
  v16h qa[4];
#pragma unroll
  for (int ks = 0; ks < 4; ++ks) {
    const _Float16* p = qbase + (size_t)n * FD_ + ks * 32;
    v8h lo = *(const v8h*)(p + 16 * g);
    v8h hi = *(const v8h*)(p + 16 * g + 8);
    qa[ks] = pack16(lo, hi);
  }
  float qn_r[8];
#pragma unroll
  for (int r = 0; r < 8; ++r) qn_r[r] = qn[(size_t)bh * N_ + i0 + r + 8 * g];

  v8f zero = {};
  float mrow[8], lrow[8];
  v8f oacc[8];
#pragma unroll
  for (int r = 0; r < 8; ++r) { mrow[r] = -3.0e38f; lrow[r] = 0.0f; }
#pragma unroll
  for (int dti = 0; dti < 8; ++dti) oacc[dti] = zero;

  const _Float16* kbase = Kr + (size_t)bh * N_ * FD_;
  const _Float16* vbase = Vt + (size_t)bh * FD_ * N_;
  int wjmax = i0 + 15;
  int nchunks = 2 * iblk + 2;     // covers j <= iblk*64+63

  // prologue: TDM loads for chunk 0 into buffer 0 (wave 0 issues)
  if (wave == 0) {
    tdm_load_2d(kbase, lds_off(&lk[0][0]), FD_, JCH_, FD_, JCH_, FD_);
    tdm_load_2d(vbase, lds_off(&lv[0][0]), JCH_, FD_, JCH_, FD_, N_);
  }
  // keep the compiler honest about TDM-written LDS
  asm volatile("" : : "r"(&lk[0][0]), "r"(&lv[0][0]) : "memory");

  for (int c = 0; c < nchunks; ++c) {
    int buf = c & 1;
    if (wave == 0) {
      if (c + 1 < nchunks) {
        size_t jn = (size_t)(c + 1) * JCH_;
        tdm_load_2d(kbase + jn * FD_, lds_off(&lk[buf ^ 1][0]),
                    FD_, JCH_, FD_, JCH_, FD_);
        tdm_load_2d(vbase + jn, lds_off(&lv[buf ^ 1][0]),
                    JCH_, FD_, JCH_, FD_, N_);
        __builtin_amdgcn_s_wait_tensorcnt((short)2);  // chunk c landed
      } else {
        __builtin_amdgcn_s_wait_tensorcnt((short)0);
      }
    }
    __syncthreads();              // publish chunk c to all 4 waves
    int j0 = c * JCH_;
    if (j0 <= wjmax) {
      const _Float16* lkb = &lk[buf][0];
      const _Float16* lvb = &lv[buf][0];
      v8f sa = zero, sb = zero;
#pragma unroll
      for (int ks = 0; ks < 4; ++ks) {
        v16h ba = *(const v16h*)(lkb + (size_t)n * FD_ + ks * 32 + 16 * g);
        v16h bb = *(const v16h*)(lkb + (size_t)(16 + n) * FD_ + ks * 32 + 16 * g);
        sa = wmma_f16(qa[ks], ba, sa);
        sb = wmma_f16(qa[ks], bb, sb);
      }
      float kn_a = kn[(size_t)bh * N_ + j0 + n];
      float kn_b = kn[(size_t)bh * N_ + j0 + 16 + n];
#pragma unroll
      for (int r = 0; r < 8; ++r) {
        int i = i0 + r + 8 * g;
        float la = -3.0e38f, lb = -3.0e38f;
        int ja = j0 + n;
        if (ja <= i) {
          float dt = (float)(i - ja);
          float e = expf(fmaxf(-mu * dt, -30.0f));
          float e2 = e * e;
          float Vv = sig2mu * (1.0f - e2) + addv;
          float maha = fmaxf(qn_r[r] + e2 * kn_a - 2.0f * e * sa[r], 0.0f) / Vv;
          la = coef * log1pf(maha * inv_nu) - 32.0f * logf(Vv);
        }
        int jb = j0 + 16 + n;
        if (jb <= i) {
          float dt = (float)(i - jb);
          float e = expf(fmaxf(-mu * dt, -30.0f));
          float e2 = e * e;
          float Vv = sig2mu * (1.0f - e2) + addv;
          float maha = fmaxf(qn_r[r] + e2 * kn_b - 2.0f * e * sb[r], 0.0f) / Vv;
          lb = coef * log1pf(maha * inv_nu) - 32.0f * logf(Vv);
        }
        float rm = fmaxf(la, lb);
#pragma unroll
        for (int off = 8; off >= 1; off >>= 1)
          rm = fmaxf(rm, __shfl_xor(rm, off, 32));
        float mn = fmaxf(mrow[r], rm);
        float scale = expf(mrow[r] - mn);
        float pa = expf(la - mn);
        float pb = expf(lb - mn);
        float rs = pa + pb;
#pragma unroll
        for (int off = 8; off >= 1; off >>= 1) rs += __shfl_xor(rs, off, 32);
        lrow[r] = lrow[r] * scale + rs;
        mrow[r] = mn;
#pragma unroll
        for (int dti = 0; dti < 8; ++dti) oacc[dti][r] *= scale;
        ldsp[wave][(r + 8 * g) * 32 + n]      = (_Float16)pa;
        ldsp[wave][(r + 8 * g) * 32 + 16 + n] = (_Float16)pb;
      }
      v16h pfrag = frag_a_lds(&ldsp[wave][0]);
#pragma unroll
      for (int dti = 0; dti < 8; ++dti) {
        v16h vf = *(const v16h*)(lvb + (size_t)(dti * 16 + n) * JCH_ + 16 * g);
        oacc[dti] = wmma_f16(pfrag, vf, oacc[dti]);
      }
    }
    __syncthreads();              // all readers done before buf^1 is refilled
  }

  float* ob = attn_out + (size_t)b * N_ * TWODK_;
#pragma unroll
  for (int r = 0; r < 8; ++r) {
    int t = i0 + r + 8 * g;
    float inv = 1.0f / lrow[r];
#pragma unroll
    for (int dti = 0; dti < 4; ++dti)
      ob[(size_t)t * TWODK_ + h * DH_ + dti * 16 + n] = oacc[dti][r] * inv;
#pragma unroll
    for (int dti = 4; dti < 8; ++dti)
      ob[(size_t)t * TWODK_ + DK_ + h * DH_ + (dti - 4) * 16 + n] =
          oacc[dti][r] * inv;
  }
}

// ---------------------------------------------------------------------------
// y = attn_out[2048,2048] @ Wo[2048,1024]
// ---------------------------------------------------------------------------
__global__ void __launch_bounds__(128) out_gemm_kernel(
    const float* __restrict__ A, const float* __restrict__ Bw,
    float* __restrict__ Y) {
  __shared__ __align__(64) _Float16 lds[4][1024];
  int wave = threadIdx.x >> 5, lane = threadIdx.x & 31;
  int tile = blockIdx.x * 4 + wave;
  int row0 = (tile >> 6) * 16;
  int c0   = (tile & 63) * 16;
  _Float16* la  = &lds[wave][0];
  _Float16* lbt = &lds[wave][512];
  int n = lane & 15, g = lane >> 4;
  v8f zero = {};
  v8f acc = zero;
  for (int k0 = 0; k0 < TWODK_; k0 += 32) {
    {
      int m = lane >> 1, hsel = lane & 1;
      const float4* p =
          (const float4*)(A + (size_t)(row0 + m) * TWODK_ + k0 + 16 * hsel);
      if (k0 + 32 < TWODK_)
        __builtin_prefetch(A + (size_t)(row0 + m) * TWODK_ + k0 + 32, 0, 1);
#pragma unroll
      for (int q = 0; q < 4; ++q) {
        float4 v = p[q];
        _Float16* d = la + m * 32 + 16 * hsel + 4 * q;
        d[0] = (_Float16)v.x; d[1] = (_Float16)v.y;
        d[2] = (_Float16)v.z; d[3] = (_Float16)v.w;
      }
    }
    {
      const float4* p = (const float4*)(Bw + (size_t)(k0 + lane) * DE_ + c0);
#pragma unroll
      for (int q = 0; q < 4; ++q) {
        float4 v = p[q];
        lbt[(4 * q + 0) * 32 + lane] = (_Float16)v.x;
        lbt[(4 * q + 1) * 32 + lane] = (_Float16)v.y;
        lbt[(4 * q + 2) * 32 + lane] = (_Float16)v.z;
        lbt[(4 * q + 3) * 32 + lane] = (_Float16)v.w;
      }
    }
    acc = wmma_f16(frag_a_lds(la), frag_b_lds(lbt), acc);
  }
#pragma unroll
  for (int r = 0; r < 8; ++r)
    Y[(size_t)(row0 + r + 8 * g) * DE_ + c0 + n] = acc[r];
}

// ---------------------------------------------------------------------------
extern "C" void kernel_launch(void* const* d_in, const int* in_sizes, int n_in,
                              void* d_out, int out_size, void* d_ws,
                              size_t ws_size, hipStream_t stream) {
  (void)in_sizes; (void)n_in; (void)out_size; (void)ws_size;
  const float* x         = (const float*)d_in[0];
  const float* Wq        = (const float*)d_in[1];
  const float* Wk        = (const float*)d_in[2];
  const float* Wv        = (const float*)d_in[3];
  const float* Wo        = (const float*)d_in[4];
  const float* omega     = (const float*)d_in[5];
  const float* mu_raw    = (const float*)d_in[6];
  const float* sigma_raw = (const float*)d_in[7];
  const float* eta_raw   = (const float*)d_in[8];
  const float* gamma_raw = (const float*)d_in[9];
  const float* tau       = (const float*)d_in[10];
  const float* nu_raw    = (const float*)d_in[11];

  char* ws = (char*)d_ws;
  const size_t sz_h = (size_t)B_ * H_ * N_ * FD_ * sizeof(_Float16);  // 8 MB
  _Float16* qr = (_Float16*)ws; ws += sz_h;
  _Float16* kr = (_Float16*)ws; ws += sz_h;
  _Float16* vT = (_Float16*)ws; ws += sz_h;
  float* qn    = (float*)ws;    ws += (size_t)B_ * H_ * N_ * sizeof(float);
  float* kn    = (float*)ws;    ws += (size_t)B_ * H_ * N_ * sizeof(float);
  float* attn  = (float*)ws;    ws += (size_t)B_ * N_ * TWODK_ * sizeof(float);

  proj_rot_kernel<<<2048, 128, 0, stream>>>(x, Wq, omega, qr, 0);
  proj_rot_kernel<<<2048, 128, 0, stream>>>(x, Wk, omega, kr, 1);
  proj_rot_kernel<<<2048, 128, 0, stream>>>(x, Wv, omega, vT, 2);
  norms_kernel<<<128, 256, 0, stream>>>(qr, kr, qn, kn);
  attn_kernel<<<512, 128, 0, stream>>>(qr, kr, vT, qn, kn, mu_raw, sigma_raw,
                                       eta_raw, gamma_raw, tau, nu_raw, attn);
  out_gemm_kernel<<<2048, 128, 0, stream>>>(attn, Wo, (float*)d_out);
}
